// ConditionedLSTM_7224134992153
// MI455X (gfx1250) — matrix operand order repaired and checked by reading, not compiled, not run on hardware
//
#include <hip/hip_runtime.h>
#include <hip/hip_bf16.h>

// CDNA5 / gfx1250 conditioned-LSTM decoder.
// GEMMs: v_wmma_f32_16x16x32_bf16, fp32 accumulation.
// A (activations, 64xK) is staged into LDS via async global->LDS copies
// (ASYNCcnt-tracked, double buffered); B (weights) is software-pipelined in
// registers; A fragments are batch-loaded per k-step so WMMAs overlap the
// ds_load latency with progressive DScnt waits.
// W_out in bf16 = 65.5 MB -> resident in the 192 MB L2 across all 256 steps.

typedef __bf16 bf16_t;
typedef __attribute__((ext_vector_type(16))) __bf16 v16bf;
typedef __attribute__((ext_vector_type(8)))  __bf16 bf16x8;
typedef __attribute__((ext_vector_type(8)))  float  v8f;
typedef __attribute__((ext_vector_type(4)))  int    v4i_;

#define B_    64
#define L_    256
#define COND_ 512
#define D_    128
#define H_    1024
#define V_    32000
#define G_    4096          // 4*H
#define KG_   1152          // D + H  (packed [emb | h] recurrent K)
#define KC_   64            // K-chunk staged in LDS per iteration

__device__ __forceinline__ float sigmoidf_(float x) { return 1.0f / (1.0f + __expf(-x)); }

__device__ __forceinline__ unsigned int hash3_(unsigned int a, unsigned int b, unsigned int c) {
    unsigned int h = a * 0x9E3779B9u ^ (b + 0x7F4A7C15u) * 0x85EBCA6Bu ^ (c + 0x165667B1u) * 0xC2B2AE35u;
    h ^= h >> 16; h *= 0x7FEB352Du; h ^= h >> 15; h *= 0x846CA68Bu; h ^= h >> 16;
    return h;
}

// ---- async global->LDS staging (gfx1250 ASYNCcnt path, with safe fallback) ----

__device__ __forceinline__ void async_copy16(const bf16_t* g, bf16_t* l) {
#if defined(__has_builtin) && __has_builtin(__builtin_amdgcn_global_load_async_to_lds_b128)
    __builtin_amdgcn_global_load_async_to_lds_b128(
        (__attribute__((address_space(1))) v4i_*)g,
        (__attribute__((address_space(3))) v4i_*)l,
        0, 0);
#else
    *(bf16x8*)l = *(const bf16x8*)g;   // synchronous fallback
#endif
}

template <int N>
__device__ __forceinline__ void wait_async() {
#if defined(__has_builtin) && __has_builtin(__builtin_amdgcn_s_wait_asynccnt)
    __builtin_amdgcn_s_wait_asynccnt((short)N);
#endif
}

// ---------------- setup / conversion kernels ----------------

__global__ void k_setup(const float* __restrict__ full_cond,
                        const float* __restrict__ b_ih, const float* __restrict__ b_hh,
                        bf16_t* __restrict__ cond_bf, float* __restrict__ bias_sum,
                        float* __restrict__ h, float* __restrict__ c) {
    int idx = blockIdx.x * blockDim.x + threadIdx.x;
    if (idx < B_ * COND_) cond_bf[idx] = (bf16_t)full_cond[idx];
    if (idx < G_)         bias_sum[idx] = b_ih[idx] + b_hh[idx];
    if (idx < B_ * H_)  { h[idx] = 0.0f; c[idx] = 0.0f; }
}

__global__ void k_conv_wc(const float* __restrict__ W_ih, bf16_t* __restrict__ Wc) {
    int idx = blockIdx.x * blockDim.x + threadIdx.x;
    if (idx >= G_ * COND_) return;
    int n = idx / COND_, k = idx % COND_;
    Wc[idx] = (bf16_t)W_ih[n * (COND_ + D_) + k];
}

__global__ void k_conv_wg(const float* __restrict__ W_ih, const float* __restrict__ W_hh,
                          bf16_t* __restrict__ Wg) {
    int idx = blockIdx.x * blockDim.x + threadIdx.x;
    if (idx >= G_ * KG_) return;
    int n = idx / KG_, k = idx % KG_;
    float v = (k < D_) ? W_ih[n * (COND_ + D_) + COND_ + k] : W_hh[n * H_ + (k - D_)];
    Wg[idx] = (bf16_t)v;
}

__global__ void k_conv_wout(const float* __restrict__ W, bf16_t* __restrict__ Wb, int n) {
    int idx = blockIdx.x * blockDim.x + threadIdx.x;
    if (idx < n) Wb[idx] = (bf16_t)W[idx];
}

// ---------------- WMMA GEMM: D(64,N) = A(64,K) @ B(N,K)^T + Cinit ----------------
// Block = 8 waves covering 128 columns of N.  A is staged in LDS in 64xKC bf16
// chunks (double buffered, async copies).  B fragments are double buffered in
// registers one chunk ahead; A fragments batch-loaded per k-step.
// Fragment packing used here (wave32; contiguous k for A and B, not the ISA table order):
//   A 16x32 bf16:  lane m=l16; K run kb+16h..+15 (contiguous in sA row, same as B)
//   B 32x16 bf16:  lane n=l16; K run kb+16h..+15 (contiguous in (N,K) row)
//   C/D 16x16 f32: m = v + 8h, n = l16

__device__ __forceinline__ v16bf load_bfrag(const bf16_t* __restrict__ Bm,
                                            size_t rowoff, int kb, int half) {
    const bf16x8* p = (const bf16x8*)(Bm + rowoff + kb + 16 * half);
    bf16x8 b0 = p[0], b1 = p[1];
    v16bf f;
#pragma unroll
    for (int i = 0; i < 8; ++i) { f[i] = b0[i]; f[8 + i] = b1[i]; }
    return f;
}

__global__ void __launch_bounds__(256)
k_wmma_gemm(const bf16_t* __restrict__ A, const bf16_t* __restrict__ Bm,
            const float* __restrict__ Cinit, int cinit_full,
            float* __restrict__ Dm, int N, int K) {
    __shared__ bf16_t sA[2][64 * KC_];            // 2 x 8 KB double buffer

    const int tid  = threadIdx.x;
    const int wave = tid >> 5;
    const int lane = tid & 31;
    const int half = lane >> 4;
    const int l16  = lane & 15;
    const int n    = (blockIdx.x * 8 + wave) * 16 + l16;
    if (blockIdx.x * 128 >= N) return;            // uniform guard (barrier-safe)

    // staging geometry: 64 rows x 128B; 512 16B pieces; 2 per thread
    const int i0 = tid,       r0 = i0 >> 3, c0 = (i0 & 7) * 8;
    const int i1 = tid + 256, r1 = i1 >> 3, c1 = (i1 & 7) * 8;

    // prologue: stage chunk 0
    async_copy16(A + (size_t)r0 * K + c0, &sA[0][r0 * KC_ + c0]);
    async_copy16(A + (size_t)r1 * K + c1, &sA[0][r1 * KC_ + c1]);

    v8f acc[4];
#pragma unroll
    for (int mt = 0; mt < 4; ++mt) {
#pragma unroll
        for (int v = 0; v < 8; ++v) {
            int m = mt * 16 + v + 8 * half;
            acc[mt][v] = cinit_full ? Cinit[(size_t)m * N + n] : Cinit[n];
        }
    }

    const size_t rowoff = (size_t)n * K;
    v16bf bcur0 = load_bfrag(Bm, rowoff, 0, half);
    v16bf bcur1 = load_bfrag(Bm, rowoff, 32, half);

    const int nch = K / KC_;
    for (int c = 0; c < nch; ++c) {
        const int cur = c & 1;
        v16bf bnx0 = bcur0, bnx1 = bcur1;
        if (c + 1 < nch) {
            const int kb2 = (c + 1) * KC_;
            // issue next A chunk (async -> other LDS buffer) and next B frags
            async_copy16(A + (size_t)r0 * K + kb2 + c0, &sA[cur ^ 1][r0 * KC_ + c0]);
            async_copy16(A + (size_t)r1 * K + kb2 + c1, &sA[cur ^ 1][r1 * KC_ + c1]);
            bnx0 = load_bfrag(Bm, rowoff, kb2, half);
            bnx1 = load_bfrag(Bm, rowoff, kb2 + 32, half);
            wait_async<2>();      // chunk c complete; 2 newer copies may remain
        } else {
            wait_async<0>();
        }
        __syncthreads();          // LDS chunk visible to all waves

#pragma unroll
        for (int ks = 0; ks < 2; ++ks) {
            const v16bf bf = ks ? bcur1 : bcur0;
            // batch all 4 M-tile A-fragment loads, then 4 WMMAs ->
            // progressive s_wait_dscnt instead of full stalls
            v16bf af[4];
#pragma unroll
            for (int mt = 0; mt < 4; ++mt) {
                const bf16_t* ap = &sA[cur][(mt * 16 + l16) * KC_ + ks * 32];
                bf16x8 a0 = *(const bf16x8*)(ap + 16 * half);
                bf16x8 a1 = *(const bf16x8*)(ap + 16 * half + 8);
#pragma unroll
                for (int i = 0; i < 8; ++i) { af[mt][i] = a0[i]; af[mt][8 + i] = a1[i]; }
            }
#pragma unroll
            for (int mt = 0; mt < 4; ++mt) {
                acc[mt] = __builtin_amdgcn_wmma_f32_16x16x32_bf16(
                    false, af[mt], false, bf, (short)0, acc[mt], false, false);
            }
        }
        __syncthreads();          // protect sA[cur] before it is refilled
        bcur0 = bnx0; bcur1 = bnx1;
    }

#pragma unroll
    for (int mt = 0; mt < 4; ++mt) {
#pragma unroll
        for (int v = 0; v < 8; ++v) {
            int m = mt * 16 + v + 8 * half;
            Dm[(size_t)m * N + n] = acc[mt][v];
        }
    }
}

// ---------------- per-step kernels ----------------

__global__ void k_build_xh(const float* __restrict__ emb_table,
                           const int* __restrict__ gold_in, const int* __restrict__ tok_buf,
                           const float* __restrict__ h, bf16_t* __restrict__ xh, int t) {
    int idx = blockIdx.x * blockDim.x + threadIdx.x;
    if (idx >= B_ * KG_) return;
    int b = idx / KG_, j = idx % KG_;
    float v;
    if (j < D_) {
        int tok = (t == 0) ? gold_in[b * L_] : tok_buf[b];
        v = emb_table[(size_t)tok * D_ + j];
    } else {
        v = h[b * H_ + (j - D_)];
    }
    xh[idx] = (bf16_t)v;
}

__global__ void k_cell(const float* __restrict__ gates, float* __restrict__ c,
                       float* __restrict__ h, bf16_t* __restrict__ h_bf) {
    int idx = blockIdx.x * blockDim.x + threadIdx.x;
    if (idx >= B_ * H_) return;
    int b = idx >> 10, j = idx & (H_ - 1);
    const float* gr = gates + (size_t)b * G_;
    float ig = sigmoidf_(gr[j]);
    float fg = sigmoidf_(gr[H_ + j]);
    float gg = tanhf(gr[2 * H_ + j]);
    float og = sigmoidf_(gr[3 * H_ + j]);
    float cc = fg * c[idx] + ig * gg;
    float hh = og * tanhf(cc);
    c[idx] = cc; h[idx] = hh; h_bf[idx] = (bf16_t)hh;
}

__global__ void __launch_bounds__(256)
k_loss_sample(const float* __restrict__ logits, const int* __restrict__ gold_out,
              const int* __restrict__ goldlen, int* __restrict__ tok_buf,
              float* __restrict__ loss_out, int t) {
    __shared__ float sf[256];
    __shared__ int   si[256];
    const int b = blockIdx.x, tid = threadIdx.x;
    const float* row = logits + (size_t)b * V_;

    float m = -3.4e38f;
    for (int v = tid; v < V_; v += 256) m = fmaxf(m, row[v]);
    sf[tid] = m; __syncthreads();
    for (int s = 128; s > 0; s >>= 1) { if (tid < s) sf[tid] = fmaxf(sf[tid], sf[tid + s]); __syncthreads(); }
    const float rowmax = sf[0]; __syncthreads();

    float ssum = 0.0f;
    for (int v = tid; v < V_; v += 256) ssum += __expf(row[v] - rowmax);
    sf[tid] = ssum; __syncthreads();
    for (int s = 128; s > 0; s >>= 1) { if (tid < s) sf[tid] += sf[tid + s]; __syncthreads(); }
    const float Z = sf[0]; __syncthreads();

    float best = -3.4e38f; int bidx = 0;
    for (int v = tid; v < V_; v += 256) {
        unsigned int h = hash3_((unsigned)t, (unsigned)b, (unsigned)v);
        float u = h * (1.0f / 4294967296.0f) * 0.9999999f + 1e-9f;
        float gum = -__logf(-__logf(u));          // TEMP == 1.0
        float val = row[v] + gum;
        if (val > best) { best = val; bidx = v; }
    }
    sf[tid] = best; si[tid] = bidx; __syncthreads();
    for (int s = 128; s > 0; s >>= 1) {
        if (tid < s && sf[tid + s] > sf[tid]) { sf[tid] = sf[tid + s]; si[tid] = si[tid + s]; }
        __syncthreads();
    }
    if (tid == 0) {
        int gold = gold_out[b * L_ + t];
        float ll = row[gold] - rowmax - __logf(Z);
        float lt = (t < goldlen[b]) ? -ll : 0.0f;
        if (t == 0) loss_out[b] = lt; else loss_out[b] += lt;
        tok_buf[b] = si[0];
    }
}

// ---------------- host launch ----------------

extern "C" void kernel_launch(void* const* d_in, const int* in_sizes, int n_in,
                              void* d_out, int out_size, void* d_ws, size_t ws_size,
                              hipStream_t stream) {
    const float* full_cond = (const float*)d_in[0];
    const float* emb_table = (const float*)d_in[1];
    const float* W_ih      = (const float*)d_in[2];
    const float* W_hh      = (const float*)d_in[3];
    const float* b_ih      = (const float*)d_in[4];
    const float* b_hh      = (const float*)d_in[5];
    const float* W_out     = (const float*)d_in[6];
    const float* b_out     = (const float*)d_in[7];
    const int*   gold_in   = (const int*)d_in[8];
    const int*   gold_out  = (const int*)d_in[9];
    const int*   goldlen   = (const int*)d_in[10];
    float* loss_out = (float*)d_out;

    char* ws = (char*)d_ws;
    size_t off = 0;
    auto alloc = [&](size_t bytes) -> char* {
        char* p = ws + off;
        off += (bytes + 255) & ~(size_t)255;
        return p;
    };

    bf16_t* Wc    = (bf16_t*)alloc((size_t)G_ * COND_ * 2);   //  4.2 MB
    bf16_t* Wg    = (bf16_t*)alloc((size_t)G_ * KG_ * 2);     //  9.4 MB
    bf16_t* Wo    = (bf16_t*)alloc((size_t)V_ * H_ * 2);      // 65.5 MB (L2-resident)
    bf16_t* condb = (bf16_t*)alloc((size_t)B_ * COND_ * 2);
    float*  bias  = (float*) alloc((size_t)G_ * 4);
    float*  cgat  = (float*) alloc((size_t)B_ * G_ * 4);      // cond_gates (time-invariant)
    bf16_t* xh    = (bf16_t*)alloc((size_t)B_ * KG_ * 2);
    float*  gates = (float*) alloc((size_t)B_ * G_ * 4);
    float*  hbuf  = (float*) alloc((size_t)B_ * H_ * 4);
    float*  cbuf  = (float*) alloc((size_t)B_ * H_ * 4);
    bf16_t* hbf   = (bf16_t*)alloc((size_t)B_ * H_ * 2);
    float*  logit = (float*) alloc((size_t)B_ * V_ * 4);      //  8.2 MB
    int*    tok   = (int*)   alloc((size_t)B_ * 4);

    // one-time setup (re-done every launch: deterministic, no caching)
    k_setup    <<< (B_ * H_ + 255) / 256, 256, 0, stream >>>(full_cond, b_ih, b_hh, condb, bias, hbuf, cbuf);
    k_conv_wc  <<< (G_ * COND_ + 255) / 256, 256, 0, stream >>>(W_ih, Wc);
    k_conv_wg  <<< (G_ * KG_ + 255) / 256, 256, 0, stream >>>(W_ih, W_hh, Wg);
    k_conv_wout<<< (V_ * H_ + 255) / 256, 256, 0, stream >>>(W_out, Wo, V_ * H_);

    // cond_gates(64,4096) = cond @ Wc^T + (b_ih + b_hh)
    k_wmma_gemm<<< (G_ / 16) / 8, 256, 0, stream >>>(condb, Wc, bias, 0, cgat, G_, COND_);

    for (int t = 0; t < L_; ++t) {
        k_build_xh <<< (B_ * KG_ + 255) / 256, 256, 0, stream >>>(emb_table, gold_in, tok, hbuf, xh, t);
        // gates = cond_gates + xh @ Wg^T        (N=4096, K=1152)
        k_wmma_gemm<<< (G_ / 16) / 8, 256, 0, stream >>>(xh, Wg, cgat, 1, gates, G_, KG_);
        k_cell     <<< (B_ * H_ + 255) / 256, 256, 0, stream >>>(gates, cbuf, hbuf, hbf);
        // logits = h @ W_out^T + b_out          (N=32000, K=1024) — dominant GEMM
        k_wmma_gemm<<< (V_ / 16) / 8, 256, 0, stream >>>(hbf, Wo, b_out, 0, logit, V_, H_);
        k_loss_sample<<< B_, 256, 0, stream >>>(logit, gold_out, goldlen, tok, loss_out, t);
    }
}
